// MyModel_12738873000510
// MI455X (gfx1250) — compile-verified, matrix operand order checked by reading, not run on hardware
//
#include <hip/hip_runtime.h>
#include <hip/hip_bf16.h>
#include <stdint.h>

// ---------------------------------------------------------------------------
// Types for CDNA5 WMMA (wave32, 16x16x32 bf16 -> f32)
// ---------------------------------------------------------------------------
typedef __attribute__((ext_vector_type(16))) __bf16 v16bf;
typedef __attribute__((ext_vector_type(8)))  float  v8f;

#define HDIM   512
#define ROWS   16
#define NTHR   256
#define BN_S   0.9999950000374997f   // 1/sqrt(1+1e-5)
#define CS_MAX 8000.0f
#define POS_MAX 100.0f

// per-MLP dwords of repacked bf16 layer-2 weights: 512*512/2
#define W2_DW_PER_MLP (HDIM*HDIM/2)

// ---------------------------------------------------------------------------
// Constant tables from the reference
// ---------------------------------------------------------------------------
__constant__ float PRESS_TAB[80] = {
  13.9f,15.9f,18.1f,20.5f,23.2f,26.2f,29.5f,33.1f,37.0f,41.3f,
  46.1f,51.2f,56.8f,62.9f,69.5f,76.7f,84.4f,92.7f,101.7f,111.3f,
  121.6f,132.7f,144.6f,157.3f,170.8f,185.2f,200.6f,216.9f,234.3f,252.7f,
  272.2f,292.8f,314.6f,337.7f,362.0f,387.6f,414.6f,443.0f,472.9f,504.3f,
  537.2f,571.7f,607.9f,645.8f,685.4f,726.9f,770.2f,815.4f,862.6f,911.8f,
  963.2f,1016.6f,1072.2f,1130.1f,1190.3f,1252.9f,1317.9f,1385.4f,1455.5f,1528.2f,
  1603.6f,1681.8f,1762.8f,1846.7f,1933.7f,2023.7f,2116.8f,2213.2f,2313.0f,2416.1f,
  2522.8f,2633.2f,2747.3f,2865.3f,2987.4f,3113.6f,3244.2f,3379.3f,3519.3f,3664.5f};
// TEM[i] = -62 + 2*i  (computed inline)

__constant__ float LO_TAB[10] = {100.f,150.f,200.f,250.f,300.f,350.f,400.f,450.f,500.f,550.f};
__constant__ float HI_TAB[10] = {200.f,400.f,600.f,800.f,1000.f,1200.f,1400.f,1600.f,1800.f,2000.f};
__constant__ float CSMIN_TAB[100] = {
  2000,2000,2000,2000,2000,2000,2000,2000,2000,2000,
  1600,1600,1600,1600,1600,1700,1800,1900,2000,2000,
  1200,1200,1200,1200,1200,1200,1200,1200,1600,2000,
   900, 900, 950,1000,1050,1100,1150,1200,1600,2000,
   800, 800, 800, 800, 900,1000,1100,1200,1600,2000,
   800, 800, 800, 800, 800, 900,1050,1200,1600,2000,
   800, 800, 800, 800, 800, 800,1000,1200,1600,2000,
   800, 800, 800, 800, 800, 800, 950,1200,1600,2000,
   800, 800, 800, 800, 800, 800, 900,1200,1600,2000,
   800, 800, 800, 800, 800, 800, 850,1200,1600,2000};

// feature vector slots per row:
// 0:n_dhi 1:ac_pid 2:n_hi 3:n_lo 4:n_t2 5:n_t5 6:n_t1 7:SCErr 8:exv 9:cs_min
__constant__ int FEAT_IDX[6][5] = {
  {0,0,0,0,0},      // cs1 <- n_dhi
  {1,0,0,0,0},      // cs2 <- ac_pid
  {2,3,4,5,0},      // cs3 <- n_hi,n_lo,n_t2,n_t5
  {7,0,0,0,0},      // cp1 <- SCErr
  {8,0,0,0,0},      // cp2 <- exv
  {2,3,4,5,6}};     // cp3 <- n_hi,n_lo,n_t2,n_t5,n_t1
__constant__ int FEAT_N[6] = {1,1,4,1,1,5};

__device__ __forceinline__ uint32_t f32_to_bf16_rne(float f) {
  uint32_t u = __float_as_uint(f);
  return (u + 0x7FFFu + ((u >> 16) & 1u)) >> 16;
}

// Fast tanh: hardware v_tanh_f32 on gfx1250 if the builtin exists, else a
// branch-free exp2/rcp form (works on -2|x| so it cannot overflow).
__device__ __forceinline__ float fast_tanhf(float x) {
#if __has_builtin(__builtin_amdgcn_tanhf)
  return __builtin_amdgcn_tanhf(x);
#else
  float ax = __builtin_fabsf(x);
  float t  = __builtin_amdgcn_exp2f(-2.8853900817779268f * ax);  // exp(-2|x|)
  float r  = (1.0f - t) * __builtin_amdgcn_rcpf(1.0f + t);
  return __builtin_copysignf(r, x);
#endif
}

// ---------------------------------------------------------------------------
// Repack kernel: layer-2 weights f32 [512,512] row-major  ->  bf16 B-fragments
// Fragment (kt,nt): 32x16 tile, stored as 256 dwords: slot = lane*8 + v,
// holding packed bf16 pair (k,k+1) with
//   k = kt*32 + (v>>2)*16 + (lane>>4)*8 + (v&3)*2 ,  n = nt*16 + (lane&15)
// ---------------------------------------------------------------------------
struct RepackArgs { const float* w[6]; uint32_t* dst; };

__global__ void repack_w2_bf16(RepackArgs a) {
  uint32_t gid = blockIdx.x * blockDim.x + threadIdx.x;
  if (gid >= 6u * W2_DW_PER_MLP) return;
  uint32_t m    = gid / W2_DW_PER_MLP;
  uint32_t rem  = gid - m * W2_DW_PER_MLP;
  uint32_t frag = rem >> 8;          // 0..511
  uint32_t slot = rem & 255u;
  uint32_t lane = slot >> 3;
  uint32_t v    = slot & 7u;
  uint32_t kt = frag >> 5, nt = frag & 31u;
  uint32_t k  = kt*32 + (v>>2)*16 + (lane>>4)*8 + (v&3)*2;
  uint32_t n  = nt*16 + (lane & 15u);
  const float* W = a.w[m];
  uint32_t lo = f32_to_bf16_rne(W[k      * HDIM + n]);
  uint32_t hi = f32_to_bf16_rne(W[(k+1u) * HDIM + n]);
  a.dst[gid] = lo | (hi << 16);
}

// ---------------------------------------------------------------------------
// Fused model kernel: 16 rows per block, 8 wave32s.
// ---------------------------------------------------------------------------
struct ModelArgs {
  const float* x;                 // [N,11]
  const float* l1w[6];            // [in,512]
  const float* l1b[6];            // [512]
  const float* l2b[6];            // [512]
  const float* l3w[6];            // [512]
  const float* l3b[6];            // [1]
  const uint32_t* l2w_tiled;      // ws: 6 * W2_DW_PER_MLP dwords (bf16 pairs)
  float* out;                     // [N,2]
};

union Frag {                      // one 16x32 bf16 WMMA operand per lane
  uint4 q[2];
  v16bf v;
};

__global__ __launch_bounds__(NTHR)
void fused_mlp6_kernel(ModelArgs P) {
  __shared__ unsigned short sh_h1[ROWS * HDIM];   // bf16 activations (A)
  __shared__ float          sh_h2[ROWS * HDIM];   // f32 activations
  __shared__ float          sh_feat[ROWS][10];
  __shared__ float          sh_part[ROWS][16];
  __shared__ float          sh_acc[2][ROWS];      // [0]=cs, [1]=cp

  const int tid  = threadIdx.x;
  const int wave = tid >> 5;
  const int lane = tid & 31;
  const int row0 = blockIdx.x * ROWS;

  // ---- Phase 0: per-row features (16 lanes) -------------------------------
  if (tid < ROWS) {
    const float* xr = P.x + (size_t)(row0 + tid) * 11;
    float ac   = xr[0];
    float lo_p = xr[1];
    float hi_p = xr[2];
    float aimh = xr[4];
    float exv  = xr[5];
    float sct  = xr[6];
    float t2   = xr[8];
    float t5   = xr[9];
    float t1   = xr[10];

    float dif_hi = aimh - hi_p;

    // tem_sat_from_press(hi_p): searchsorted-left on PRESS_TAB
    int s = 0;
    #pragma unroll
    for (int i = 0; i < 80; ++i) s += (PRESS_TAB[i] < hi_p) ? 1 : 0;
    int idx = min(max(s - 1, 0), 78);
    float p1 = PRESS_TAB[idx], p2 = PRESS_TAB[idx + 1];
    float tA = -62.0f + 2.0f * idx, tB = tA + 2.0f;
    float w  = (hi_p - p1) / (p2 - p1);
    float hi_sat = w * (tB - tA) + tA;

    float scerr = hi_sat - t5 - sct;

    // bilinear cs_min
    int sl = 0, sh = 0;
    #pragma unroll
    for (int i = 0; i < 10; ++i) { sl += (LO_TAB[i] < lo_p) ? 1 : 0; sh += (HI_TAB[i] < hi_p) ? 1 : 0; }
    int li = min(max(sl - 1, 0), 8);
    int hi = min(max(sh - 1, 0), 8);
    float Q11 = CSMIN_TAB[li*10 + hi],     Q12 = CSMIN_TAB[li*10 + hi + 1];
    float Q21 = CSMIN_TAB[(li+1)*10 + hi], Q22 = CSMIN_TAB[(li+1)*10 + hi + 1];
    float xr_ = (lo_p - LO_TAB[li]) / (LO_TAB[li+1] - LO_TAB[li]);
    float yr_ = (hi_p - HI_TAB[hi]) / (HI_TAB[hi+1] - HI_TAB[hi]);
    float R1 = xr_ * (Q21 - Q11) + Q11;
    float R2 = xr_ * (Q22 - Q12) + Q12;
    float cs_min = yr_ * (R2 - R1) + R1;

    sh_feat[tid][0] = (dif_hi + 2000.0f) * (1.0f / 4000.0f);
    sh_feat[tid][1] = ac;
    sh_feat[tid][2] = (hi_p - 900.0f) * (1.0f / 1500.0f);
    sh_feat[tid][3] = (lo_p - 100.0f) * (1.0f / 600.0f);
    sh_feat[tid][4] = (t2 - 40.0f) * (1.0f / 75.0f);
    sh_feat[tid][5] = (t5 - 20.0f) * (1.0f / 45.0f);
    sh_feat[tid][6] = (t1 + 18.0f) * (1.0f / 49.0f);
    sh_feat[tid][7] = scerr;
    sh_feat[tid][8] = exv;
    sh_feat[tid][9] = cs_min;
    sh_acc[0][tid] = 0.0f;
    sh_acc[1][tid] = 0.0f;
  }
  __syncthreads();

  const uint32_t* h1u = (const uint32_t*)sh_h1;

  // ---- Per-MLP fused pipeline --------------------------------------------
  for (int j = 0; j < 6; ++j) {
    // Layer 1: [16,in] @ [in,512] -> tanh -> bf16 LDS
    {
      const int row = tid & 15;
      const int c0  = tid >> 4;            // 0..15, stride 16 over cols
      const int nin = FEAT_N[j];
      float f[5];
      #pragma unroll
      for (int i = 0; i < 5; ++i) f[i] = (i < nin) ? sh_feat[row][FEAT_IDX[j][i]] : 0.0f;
      const float* W1 = P.l1w[j];
      const float* B1 = P.l1b[j];
      #pragma unroll 4
      for (int c = c0; c < HDIM; c += 16) {
        float acc = B1[c];
        #pragma unroll
        for (int i = 0; i < 5; ++i) if (i < nin) acc += f[i] * W1[i * HDIM + c];
        float h = fast_tanhf(acc * BN_S);
        sh_h1[row * HDIM + c] = (unsigned short)f32_to_bf16_rne(h);
      }
    }
    __syncthreads();

    // Layer 2: WMMA bf16 [16,512] @ [512,512] -> tanh -> f32 LDS
    // Each wave owns 4 N-tiles (wave, wave+8, wave+16, wave+24) with 4
    // accumulators; A fragments are loaded once per k-step and feed all 4
    // WMMAs; A/B fragments are double-buffered so loads for k+1 are in
    // flight behind the 4 WMMAs of k.
    {
      const uint32_t* w2 = P.l2w_tiled + (size_t)j * W2_DW_PER_MLP;
      const float*    B2 = P.l2b[j];
      const int m    = lane & 15;
      const int half = lane >> 4;
      // A dword index = m*256 + kt*16 + (v>>2)*8 + half*4 + (v&3)
      const uint32_t* aP = h1u + (m << 8) + (half << 2);

      Frag A[2];
      Frag B[2][4];
      v8f  acc[4];
      #pragma unroll
      for (int t = 0; t < 4; ++t)
        acc[t] = (v8f){0.f,0.f,0.f,0.f,0.f,0.f,0.f,0.f};

      // preload kt = 0
      A[0].q[0] = *(const uint4*)(aP);
      A[0].q[1] = *(const uint4*)(aP + 8);
      #pragma unroll
      for (int t = 0; t < 4; ++t) {
        const uint4* bp =
          (const uint4*)(w2 + ((size_t)(wave + t * 8) << 8) + (lane << 3));
        B[0][t].q[0] = bp[0];
        B[0][t].q[1] = bp[1];
      }

      #pragma unroll
      for (int kt = 0; kt < 16; ++kt) {
        const int cur = kt & 1;
        const int nxt = cur ^ 1;
        if (kt < 15) {
          const uint32_t* ap = aP + (kt + 1) * 16;
          A[nxt].q[0] = *(const uint4*)(ap);
          A[nxt].q[1] = *(const uint4*)(ap + 8);
          #pragma unroll
          for (int t = 0; t < 4; ++t) {
            const uint4* bp = (const uint4*)(
              w2 + ((size_t)((kt + 1) * 32 + wave + t * 8) << 8) + (lane << 3));
            B[nxt][t].q[0] = bp[0];
            B[nxt][t].q[1] = bp[1];
          }
        }
        #pragma unroll
        for (int t = 0; t < 4; ++t)
          acc[t] = __builtin_amdgcn_wmma_f32_16x16x32_bf16(
                     false, A[cur].v, false, B[cur][t].v, (short)0, acc[t],
                     false, false);
      }

      // epilogue: bias + tanh -> f32 LDS
      #pragma unroll
      for (int t = 0; t < 4; ++t) {
        const int n = (wave + t * 8) * 16 + m;
        const float bias = B2[n];
        #pragma unroll
        for (int v = 0; v < 8; ++v) {
          const int mm = v + 8 * half;
          sh_h2[mm * HDIM + n] = fast_tanhf((acc[t][v] + bias) * BN_S);
        }
      }
    }
    __syncthreads();

    // Layer 3: [16,512] . [512] + b  (partial sums + LDS reduce)
    {
      const int row = tid >> 4;
      const int seg = tid & 15;
      const float* W3 = P.l3w[j];
      float s = 0.0f;
      #pragma unroll 8
      for (int i = 0; i < 32; ++i) {
        const int c = seg * 32 + i;
        s += sh_h2[row * HDIM + c] * W3[c];
      }
      sh_part[row][seg] = s;
    }
    __syncthreads();
    if (tid < ROWS) {
      float tot = P.l3b[j][0];
      #pragma unroll
      for (int seg = 0; seg < 16; ++seg) tot += sh_part[tid][seg];
      sh_acc[(j >= 3) ? 1 : 0][tid] += tot;
    }
    __syncthreads();
  }

  // ---- Final clamp + store ------------------------------------------------
  if (tid < ROWS) {
    float cs = sh_acc[0][tid] * CS_MAX;
    float cp = sh_acc[1][tid] * POS_MAX;
    float cs_min = sh_feat[tid][9];
    float speed = fminf(fmaxf(cs, cs_min), CS_MAX);
    float pos   = fminf(fmaxf(cp, 0.0f), POS_MAX);
    float* o = P.out + (size_t)(row0 + tid) * 2;
    o[0] = speed;
    o[1] = pos;
  }
}

// ---------------------------------------------------------------------------
// Launch
// ---------------------------------------------------------------------------
extern "C" void kernel_launch(void* const* d_in, const int* in_sizes, int n_in,
                              void* d_out, int out_size, void* d_ws, size_t ws_size,
                              hipStream_t stream) {
  // Inputs: [0]=x, then 6 groups of {W0,W1,W2,b0,b1,b2}.
  // Detect flattening order of the param dict: group 2's W0 is 4*512 for
  // insertion order (cs1,cs2,cs3,cp1,cp2,cp3) or 5*512 for sorted keys
  // (cp1,cp2,cp3,cs1,cs2,cs3).
  const int g2w0 = in_sizes[1 + 2 * 6 + 0];
  const bool insertion_order = (g2w0 == 4 * 512);

  // canonical order: cs1, cs2, cs3, cp1, cp2, cp3
  int grp[6];
  for (int c = 0; c < 6; ++c) grp[c] = insertion_order ? c : ((c + 3) % 6);

  ModelArgs M;
  RepackArgs R;
  M.x = (const float*)d_in[0];
  for (int c = 0; c < 6; ++c) {
    const int base = 1 + grp[c] * 6;
    M.l1w[c] = (const float*)d_in[base + 0];
    R.w[c]   = (const float*)d_in[base + 1];   // layer-2 weights, to repack
    M.l3w[c] = (const float*)d_in[base + 2];
    M.l1b[c] = (const float*)d_in[base + 3];
    M.l2b[c] = (const float*)d_in[base + 4];
    M.l3b[c] = (const float*)d_in[base + 5];
  }
  R.dst       = (uint32_t*)d_ws;
  M.l2w_tiled = (const uint32_t*)d_ws;
  M.out       = (float*)d_out;

  // 1) Repack 6x 512x512 f32 weights -> swizzled bf16 B-fragments in ws.
  {
    const uint32_t total = 6u * W2_DW_PER_MLP;
    repack_w2_bf16<<<(total + NTHR - 1) / NTHR, NTHR, 0, stream>>>(R);
  }

  // 2) Fused 6-MLP forward, 16 rows per block.
  {
    const int nrows  = in_sizes[0] / 11;
    const int blocks = nrows / ROWS;
    fused_mlp6_kernel<<<blocks, NTHR, 0, stream>>>(M);
  }
}
